// MetaModel_46420006535850
// MI455X (gfx1250) — matrix operand order repaired and checked by reading, not compiled, not run on hardware
//
#include <hip/hip_runtime.h>
#include <hip/hip_bf16.h>
#include <math.h>

// ---------------------------------------------------------------------------
// MetaModel hypernetwork, restructured for MI455X (gfx1250, wave32, WMMA).
//
// W1(x) = z1(x) @ Wf + bf with z1 in R^5  =>  the 629MB batched weight tensor
// collapses into dense GEMMs:
//   P1[n, j*200+h] = (X @ Wf_j)[n,h]  (j=0..4),  P1[n,1000+h] = (X @ bf)[n,h]
//   mid = tanh( sum_j z1[n,j]*P1[n,j*200+h] + P1[n,1000+h] + b1(n,h) )
// (same for layer 2).  Heavy GEMMs: bf16 WMMA, f32 accumulate.
//
// Data movement (CDNA5 paths):
//  - A tile: TENSOR_LOAD_TO_LDS (TDM, one op per k-step from wave 0; D# pad
//    feature generates the 80B LDS row stride in hardware), TENSORcnt-gated.
//  - B tile: GLOBAL_LOAD_ASYNC_TO_LDS_B128 per lane, ASYNCcnt-gated.
//  - manual double buffering; last iteration peeled so the hot loop is
//    branch-free (EXEC all-ones around v_wmma as required).
// ---------------------------------------------------------------------------

typedef __attribute__((ext_vector_type(16))) __bf16 v16bf;
typedef __attribute__((ext_vector_type(8)))  float  v8f;
typedef __attribute__((ext_vector_type(4)))  unsigned uv4;
typedef __attribute__((ext_vector_type(8)))  int      iv8;
typedef __attribute__((ext_vector_type(4)))  int      iv4;

#define NI      1024      // batch
#define CONTENT 768
#define EMBD    200
#define HD      200

// GEMM tiling: 256 threads = 8 waves, 64x128 block tile, 32x32 per wave.
#define TILE_M 64
#define TILE_N 128
#define KSTEP  32
#define LDSA_S 40         // 32 bf16 + 8 pad -> 80B row stride, 16B aligned
#define LDSB_S 40

// --- CDNA5 async global->LDS copy (16B per lane), inline asm ---------------
// VDST = LDS byte address (low 32 bits of the generic pointer ARE the LDS
// offset on gfx1250: flat->LDS truncates to addr[31:0], ISA 10.2).
#define ASYNC_LDS_B128(ldsPtr, gPtr)                                          \
    asm volatile("global_load_async_to_lds_b128 %0, %1, off"                  \
                 :: "v"((unsigned)(uintptr_t)(ldsPtr)), "v"(gPtr) : "memory")

#if __has_builtin(__builtin_amdgcn_s_wait_asynccnt)
#define WAIT_ASYNC(n) __builtin_amdgcn_s_wait_asynccnt(n)
#else
#define WAIT_ASYNC(n) asm volatile("s_wait_asynccnt %0" :: "i"(n) : "memory")
#endif

#if __has_builtin(__builtin_amdgcn_tensor_load_to_lds) && \
    __has_builtin(__builtin_amdgcn_s_wait_tensorcnt)
#define USE_TDM_A 1
#else
#define USE_TDM_A 0
#endif

// ---------------------------------------------------------------------------
// Generic bf16 WMMA GEMM:  C[M x ncolsPad] = A[M x K] * B^T, B stored
// column-major as BT[col][K] (ldb = Kpad).  K / M / ncolsPad are multiples of
// 32 / 64 / 128 (zero-padded in workspace) -> no guards.
// ---------------------------------------------------------------------------
__global__ __launch_bounds__(256)
void gemm_bf16_wmma(const __bf16* __restrict__ A, int lda,
                    const __bf16* __restrict__ B, int ldb,
                    float* __restrict__ C, int ldc, int K)
{
    __shared__ __align__(16) __bf16 sA[2][TILE_M * LDSA_S];
    __shared__ __align__(16) __bf16 sB[2][TILE_N * LDSB_S];

    const int tid  = threadIdx.x;
    const int lane = tid & 31;
    const int wave = tid >> 5;
    const int wr   = wave >> 2;           // 0..1  (wave row)
    const int wc   = wave & 3;            // 0..3  (wave col)
    const int m0   = blockIdx.y * TILE_M;
    const int n0   = blockIdx.x * TILE_N;

    const int l15 = lane & 15;
    const int lh  = lane >> 4;            // half-wave select

    // fragment LDS offsets (elements), per documented 16-bit layouts:
    // A (16x32): lanes0-15 row=lane K{0..7,16..23}; lanes16-31 K{8..15,24..31}
    const int aoff0 = (wr * 32 + l15) * LDSA_S + lh * 8;
    const int aoff1 = aoff0 + 16 * LDSA_S;
    // B (32x16): lanes0-15 col=lane K{0..7,16..23}; lanes16-31 K{8..15,24..31}
    const int boff0 = (wc * 32 + l15) * LDSB_S + lh * 8;
    const int boff1 = boff0 + 16 * LDSB_S;

    v8f c00 = {}, c01 = {}, c10 = {}, c11 = {};

    // ---- B tile staging: 128 cols x 32 k, two 16B async copies per thread
    auto stageB = [&](int buf, int k0) {
#pragma unroll
        for (int i = 0; i < 2; ++i) {
            int id   = tid + i * 256;
            int col  = id >> 2;
            int koff = (id & 3) * 8;
            ASYNC_LDS_B128(&sB[buf][col * LDSB_S + koff],
                           B + (size_t)(n0 + col) * ldb + k0 + koff);
        }
    };

#if USE_TDM_A
    // ---- A tile via Tensor Data Mover: 2D tile 32(k) x 64(rows), bf16.
    // D# per ISA 8.3-8.5: count=1, type=2; data_size=1(2B); pad_enable with
    // pad_interval=3 (16 DWORDs = one 64B row) and pad_amount=3 (4 DWORDs =
    // 16B) -> hardware produces the 80B padded LDS row stride (LDSA_S).
    auto tdmA = [&](int buf, int k0) {
        unsigned long long ga =
            (unsigned long long)(uintptr_t)(A + (size_t)m0 * lda + k0);
        unsigned lds = (unsigned)(uintptr_t)&sA[buf][0];
        uv4 g0 = { 1u,                                   // count=1 valid D#
                   lds,                                  // lds_addr (bytes)
                   (unsigned)ga,                         // global_addr lo
                   (unsigned)(ga >> 32) | 0x80000000u }; // addr hi | type=2
        iv8 g1 = { (int)0x06D10000,        // data_size=2B, pad_en, int=16dw, amt=4dw
                   (int)(32u << 16),       // tensor_dim0 = 32 (lo16 in [63:48])
                   (int)(64u << 16),       // tensor_dim0 hi=0 | tensor_dim1 = 64
                   (int)(32u << 16),       // tensor_dim1 hi=0 | tile_dim0 = 32
                   64,                     // tile_dim1 = 64, tile_dim2 = 0
                   lda,                    // tensor_dim0_stride (elements)
                   0, 0 };
        iv4 z4 = { 0, 0, 0, 0 };
        iv8 z8 = { 0, 0, 0, 0, 0, 0, 0, 0 };
        // clang-23 / therock 6-arg form: (g0, g1, g2, g3, extra, cpol)
        __builtin_amdgcn_tensor_load_to_lds(g0, g1, z4, z4, z8, 0);
    };
    auto stageA = [&](int buf, int k0) { if (wave == 0) tdmA(buf, k0); };
#define WAIT_TILE(nA, nB)                                           \
    do { if (wave == 0) __builtin_amdgcn_s_wait_tensorcnt(nA);      \
         WAIT_ASYNC(nB); } while (0)
#else
    // ---- fallback: A tile per-lane async (one 16B chunk per thread)
    const int arow  = tid >> 2;
    const int akoff = (tid & 3) * 8;
    auto stageA = [&](int buf, int k0) {
        ASYNC_LDS_B128(&sA[buf][arow * LDSA_S + akoff],
                       A + (size_t)(m0 + arow) * lda + k0 + akoff);
    };
#define WAIT_TILE(nA, nB) WAIT_ASYNC((nA) + (nB))
#endif

    // ---- compute one 32-k step from buffer `buf` ----
    auto compute = [&](int buf) {
        const __bf16* pa = sA[buf];
        const __bf16* pb = sB[buf];
        v16bf a0, a1, b0, b1;
        ((uint4*)&a0)[0] = *(const uint4*)&pa[aoff0];
        ((uint4*)&a0)[1] = *(const uint4*)&pa[aoff0 + 16];
        ((uint4*)&a1)[0] = *(const uint4*)&pa[aoff1];
        ((uint4*)&a1)[1] = *(const uint4*)&pa[aoff1 + 16];
        ((uint4*)&b0)[0] = *(const uint4*)&pb[boff0];
        ((uint4*)&b0)[1] = *(const uint4*)&pb[boff0 + 16];
        ((uint4*)&b1)[0] = *(const uint4*)&pb[boff1];
        ((uint4*)&b1)[1] = *(const uint4*)&pb[boff1 + 16];
        c00 = __builtin_amdgcn_wmma_f32_16x16x32_bf16(false, a0, false, b0, (short)0, c00, false, false);
        c01 = __builtin_amdgcn_wmma_f32_16x16x32_bf16(false, a0, false, b1, (short)0, c01, false, false);
        c10 = __builtin_amdgcn_wmma_f32_16x16x32_bf16(false, a1, false, b0, (short)0, c10, false, false);
        c11 = __builtin_amdgcn_wmma_f32_16x16x32_bf16(false, a1, false, b1, (short)0, c11, false, false);
    };

    // ---- double-buffered pipeline, last iteration peeled ----
    stageA(0, 0);
    stageB(0, 0);
    int cur = 0, k0 = 0;
    for (; k0 + KSTEP < K; k0 += KSTEP) {
        stageA(cur ^ 1, k0 + KSTEP);      // tile t+1 in flight
        stageB(cur ^ 1, k0 + KSTEP);
        WAIT_TILE(1, 2);                  // tile t complete (in-order cnts)
        __syncthreads();
        compute(cur);
        __syncthreads();                  // buf free before t+2 overwrites it
        cur ^= 1;
    }
    WAIT_TILE(0, 0);
    __syncthreads();
    compute(cur);

    // C/D layout: VGPR i -> M = lh*8 + i, N = lane&15
    const int rb = m0 + wr * 32 + lh * 8;
    const int cb = n0 + wc * 32 + l15;
#pragma unroll
    for (int i = 0; i < 8; ++i) {
        C[(size_t)(rb + i)      * ldc + cb]      = c00[i];
        C[(size_t)(rb + i)      * ldc + cb + 16] = c01[i];
        C[(size_t)(rb + 16 + i) * ldc + cb]      = c10[i];
        C[(size_t)(rb + 16 + i) * ldc + cb + 16] = c11[i];
    }
}

// ---------------------------------------------------------------------------
// f32 -> bf16 convert with K zero-padding: dst[r][Kpad]
// ---------------------------------------------------------------------------
__global__ void convert_pad(const float* __restrict__ src, __bf16* __restrict__ dst,
                            int K, int Kpad)
{
    int idx = blockIdx.x * blockDim.x + threadIdx.x;
    int r = idx / Kpad, k = idx - r * Kpad;
    float v = (k < K) ? src[(size_t)r * K + k] : 0.f;
    dst[idx] = (__bf16)v;
}

// ---------------------------------------------------------------------------
// Build transposed bf16 weight panel BT[colsPad][Kpad]:
//  col < colsMain : srcMain[(col/segW)*segStride + col%segW + k*ldk]
//  else           : srcBias[(col-colsMain)                + k*ldk]
// zero outside K / colsTot.
// ---------------------------------------------------------------------------
__global__ void build_bt(const float* __restrict__ srcMain,
                         const float* __restrict__ srcBias,
                         __bf16* __restrict__ dst,
                         int K, int Kpad, int colsMain, int segW,
                         long segStride, int colsTot, int ldk)
{
    int idx = blockIdx.x * blockDim.x + threadIdx.x;
    int col = idx / Kpad, k = idx - col * Kpad;
    float v = 0.f;
    if (k < K && col < colsTot) {
        if (col < colsMain)
            v = srcMain[(size_t)(col / segW) * segStride + (col % segW) + (size_t)k * ldk];
        else
            v = srcBias[(col - colsMain) + (size_t)k * ldk];
    }
    dst[idx] = (__bf16)v;
}

// ---------------------------------------------------------------------------
// Four tiny hypernetwork trunks 768->32(relu)->5(relu), one block per item.
// ---------------------------------------------------------------------------
struct HidParams {
    const float* w0[4]; const float* b0[4];
    const float* w1[4]; const float* b1[4];
    float* z[4];                       // each [N][8] (j<5 valid)
};

__global__ __launch_bounds__(128)
void hidden_mlp(const float* __restrict__ X, HidParams p)
{
    __shared__ float sx[CONTENT];
    __shared__ float sh[128];
    const int n = blockIdx.x;
    const int t = threadIdx.x;
    const float* xr = X + (size_t)n * CONTENT;
    for (int i = t; i < CONTENT; i += 128) sx[i] = xr[i];
    __syncthreads();
    {
        int g = t >> 5, u = t & 31;
        const float* w = p.w0[g];
        float acc = p.b0[g][u];
        for (int c = 0; c < CONTENT; ++c) acc = fmaf(sx[c], w[c * 32 + u], acc);
        sh[t] = fmaxf(acc, 0.f);
    }
    __syncthreads();
    if (t < 32) {
        int g = t >> 3, j = t & 7;
        if (j < 5) {
            const float* w = p.w1[g];
            float acc = p.b1[g][j];
#pragma unroll
            for (int u = 0; u < 32; ++u) acc = fmaf(sh[g * 32 + u], w[u * 5 + j], acc);
            p.z[g][(size_t)n * 8 + j] = fmaxf(acc, 0.f);
        }
    }
}

// ---------------------------------------------------------------------------
// Combines (fold 5-term z-contraction + generated bias, then tanh / output)
// ---------------------------------------------------------------------------
__global__ void combine_mid(const float* __restrict__ P,   // [N][1280]
                            const float* __restrict__ z,   // [N][8]
                            const float* __restrict__ zb,  // [N][8]
                            const float* __restrict__ Wfb, // [5][200]
                            const float* __restrict__ bfb, // [200]
                            __bf16* __restrict__ dst)      // [N][224]
{
    int idx = blockIdx.x * blockDim.x + threadIdx.x;       // N*224
    int n = idx / 224, h = idx - n * 224;
    float v = 0.f;
    if (h < HD) {
        const float* p = P + (size_t)n * 1280;
        float acc = p[1000 + h];
#pragma unroll
        for (int j = 0; j < 5; ++j) acc += z[n * 8 + j] * p[j * 200 + h];
        float b = bfb[h];
#pragma unroll
        for (int j = 0; j < 5; ++j) b += zb[n * 8 + j] * Wfb[j * 200 + h];
        v = tanhf(acc + b);
    }
    dst[idx] = (__bf16)v;
}

__global__ void combine_item(const float* __restrict__ P, const float* __restrict__ z,
                             const float* __restrict__ zb, const float* __restrict__ Wfb,
                             const float* __restrict__ bfb, float* __restrict__ out)
{
    int idx = blockIdx.x * blockDim.x + threadIdx.x;       // N*200
    int n = idx / HD, o = idx - n * HD;
    const float* p = P + (size_t)n * 1280;
    float acc = p[1000 + o];
#pragma unroll
    for (int j = 0; j < 5; ++j) acc += z[n * 8 + j] * p[j * 200 + o];
    float b = bfb[o];
#pragma unroll
    for (int j = 0; j < 5; ++j) b += zb[n * 8 + j] * Wfb[j * 200 + o];
    out[idx] = acc + b;
}

__global__ void combine_user_mid(const float* __restrict__ Pu, const float* __restrict__ bu,
                                 __bf16* __restrict__ dst)
{
    int idx = blockIdx.x * blockDim.x + threadIdx.x;       // N*224
    int n = idx / 224, h = idx - n * 224;
    float v = 0.f;
    if (h < HD) v = tanhf(Pu[(size_t)n * 256 + h] + bu[h]);
    dst[idx] = (__bf16)v;
}

__global__ void combine_user_out(const float* __restrict__ Pu, const float* __restrict__ bu,
                                 float* __restrict__ out)
{
    int idx = blockIdx.x * blockDim.x + threadIdx.x;       // N*200
    int n = idx / HD, o = idx - n * HD;
    out[idx] = Pu[(size_t)n * 256 + o] + bu[o];
}

// ---------------------------------------------------------------------------
extern "C" void kernel_launch(void* const* d_in, const int* in_sizes, int n_in,
                              void* d_out, int out_size, void* d_ws, size_t ws_size,
                              hipStream_t stream)
{
    (void)in_sizes; (void)n_in; (void)out_size; (void)ws_size;
    // setup_inputs() insertion order, params flattened recursively:
    const float* user_emb = (const float*)d_in[0];   // [1024,200]
    const float* xc       = (const float*)d_in[1];   // [1024,768]
    const float* g1w_W0 = (const float*)d_in[2];  const float* g1w_b0 = (const float*)d_in[3];
    const float* g1w_W1 = (const float*)d_in[4];  const float* g1w_b1 = (const float*)d_in[5];
    const float* g1w_W2 = (const float*)d_in[6];  const float* g1w_b2 = (const float*)d_in[7];   // [5,153600],[153600]
    const float* g1b_W0 = (const float*)d_in[8];  const float* g1b_b0 = (const float*)d_in[9];
    const float* g1b_W1 = (const float*)d_in[10]; const float* g1b_b1 = (const float*)d_in[11];
    const float* g1b_W2 = (const float*)d_in[12]; const float* g1b_b2 = (const float*)d_in[13];  // [5,200],[200]
    const float* g2w_W0 = (const float*)d_in[14]; const float* g2w_b0 = (const float*)d_in[15];
    const float* g2w_W1 = (const float*)d_in[16]; const float* g2w_b1 = (const float*)d_in[17];
    const float* g2w_W2 = (const float*)d_in[18]; const float* g2w_b2 = (const float*)d_in[19];  // [5,40000],[40000]
    const float* g2b_W0 = (const float*)d_in[20]; const float* g2b_b0 = (const float*)d_in[21];
    const float* g2b_W1 = (const float*)d_in[22]; const float* g2b_b1 = (const float*)d_in[23];
    const float* g2b_W2 = (const float*)d_in[24]; const float* g2b_b2 = (const float*)d_in[25];  // [5,200],[200]
    const float* wu1 = (const float*)d_in[26];    const float* bu1 = (const float*)d_in[27];
    const float* wu2 = (const float*)d_in[28];    const float* bu2 = (const float*)d_in[29];

    float* outUser = (float*)d_out;                  // [1024,200]
    float* outItem = (float*)d_out + (size_t)NI * HD;

    // ---- workspace carve-up (~12.2 MB total) ----
    char* ws = (char*)d_ws;  size_t off = 0;
    auto take = [&](size_t bytes) { char* p = ws + off; off += (bytes + 255) & ~(size_t)255; return p; };
    __bf16* Xbf  = (__bf16*)take((size_t)NI * 768 * 2);
    __bf16* Ubf  = (__bf16*)take((size_t)NI * 224 * 2);
    __bf16* Mbf  = (__bf16*)take((size_t)NI * 224 * 2);
    __bf16* U1bf = (__bf16*)take((size_t)NI * 224 * 2);
    __bf16* BT1  = (__bf16*)take((size_t)1280 * 768 * 2);
    __bf16* BT2  = (__bf16*)take((size_t)1280 * 224 * 2);
    __bf16* BTu1 = (__bf16*)take((size_t)256 * 224 * 2);
    __bf16* BTu2 = (__bf16*)take((size_t)256 * 224 * 2);
    float*  Z1   = (float*)take((size_t)NI * 8 * 4);
    float*  Z1b  = (float*)take((size_t)NI * 8 * 4);
    float*  Z2   = (float*)take((size_t)NI * 8 * 4);
    float*  Z2b  = (float*)take((size_t)NI * 8 * 4);
    float*  P    = (float*)take((size_t)NI * 1280 * 4);   // reused for both item GEMMs
    float*  Pu   = (float*)take((size_t)NI * 256 * 4);    // reused for both user GEMMs

    // ---- prep: bf16 conversion + transposed weight panels ----
    convert_pad<<<dim3((NI * 768) / 256), 256, 0, stream>>>(xc, Xbf, 768, 768);
    convert_pad<<<dim3((NI * 224) / 256), 256, 0, stream>>>(user_emb, Ubf, 200, 224);
    build_bt<<<dim3((1280 * 768) / 256), 256, 0, stream>>>(g1w_W2, g1w_b2, BT1, 768, 768, 1000, 200, 153600L, 1200, 200);
    build_bt<<<dim3((1280 * 224) / 256), 256, 0, stream>>>(g2w_W2, g2w_b2, BT2, 200, 224, 1000, 200, 40000L, 1200, 200);
    build_bt<<<dim3((256 * 224) / 256), 256, 0, stream>>>(wu1, wu1, BTu1, 200, 224, 200, 200, 0L, 200, 200);
    build_bt<<<dim3((256 * 224) / 256), 256, 0, stream>>>(wu2, wu2, BTu2, 200, 224, 200, 200, 0L, 200, 200);

    // ---- tiny trunks -> z codes ----
    HidParams hp;
    hp.w0[0] = g1w_W0; hp.b0[0] = g1w_b0; hp.w1[0] = g1w_W1; hp.b1[0] = g1w_b1; hp.z[0] = Z1;
    hp.w0[1] = g1b_W0; hp.b0[1] = g1b_b0; hp.w1[1] = g1b_W1; hp.b1[1] = g1b_b1; hp.z[1] = Z1b;
    hp.w0[2] = g2w_W0; hp.b0[2] = g2w_b0; hp.w1[2] = g2w_W1; hp.b1[2] = g2w_b1; hp.z[2] = Z2;
    hp.w0[3] = g2b_W0; hp.b0[3] = g2b_b0; hp.w1[3] = g2b_W1; hp.b1[3] = g2b_b1; hp.z[3] = Z2b;
    hidden_mlp<<<dim3(NI), 128, 0, stream>>>(xc, hp);

    // ---- item branch ----
    gemm_bf16_wmma<<<dim3(1280 / TILE_N, NI / TILE_M), 256, 0, stream>>>(Xbf, 768, BT1, 768, P, 1280, 768);
    combine_mid<<<dim3((NI * 224) / 256), 256, 0, stream>>>(P, Z1, Z1b, g1b_W2, g1b_b2, Mbf);
    gemm_bf16_wmma<<<dim3(1280 / TILE_N, NI / TILE_M), 256, 0, stream>>>(Mbf, 224, BT2, 224, P, 1280, 224);
    combine_item<<<dim3((NI * HD) / 256), 256, 0, stream>>>(P, Z2, Z2b, g2b_W2, g2b_b2, outItem);

    // ---- user branch ----
    gemm_bf16_wmma<<<dim3(256 / TILE_N, NI / TILE_M), 256, 0, stream>>>(Ubf, 224, BTu1, 224, Pu, 256, 224);
    combine_user_mid<<<dim3((NI * 224) / 256), 256, 0, stream>>>(Pu, bu1, U1bf);
    gemm_bf16_wmma<<<dim3(256 / TILE_N, NI / TILE_M), 256, 0, stream>>>(U1bf, 224, BTu2, 224, Pu, 256, 224);
    combine_user_out<<<dim3((NI * HD) / 256), 256, 0, stream>>>(Pu, bu2, outUser);
}
